// Stage_9165460210160
// MI455X (gfx1250) — matrix operand order repaired and checked by reading, not compiled, not run on hardware
//
#include <hip/hip_runtime.h>
#include <hip/hip_bf16.h>
#include <math.h>

// ---------------- problem constants ----------------
#define N_PTS   16384
#define KNN_K   16
#define DIM     256
#define HIDD    512
#define DIN     512
#define DSTATE  64
#define NHEADS  8
#define HEADD   64
#define ZCOLS   1160          // 2*512 + 2*64 + 8
#define ZLD     1168          // f32 zx pitch (multiple of 16)
#define XBCC    640           // 512 + 2*64
#define LCHUNK  128
#define NCHUNK  (N_PTS / LCHUNK)

typedef __bf16 bf16;
typedef __attribute__((ext_vector_type(16))) __bf16 v16bf;
typedef __attribute__((ext_vector_type(8)))  float  v8f;

__device__ __forceinline__ float gelu_f(float x) {
  const float c = 0.7978845608028654f;
  return 0.5f * x * (1.f + tanhf(c * (x + 0.044715f * x * x * x)));
}
__device__ __forceinline__ float silu_f(float x) { return x / (1.f + __expf(-x)); }
__device__ __forceinline__ bf16 to_bf16(float f) { return (bf16)f; }

// Store one f32 value as bf16 into a packed-A (WMMA fragment-linear) buffer.
// Packed-A layout: u32[(mt*KT + kt)*256 + lane*8 + v], lane = (m&15)|(half<<4),
// dword v holds k_local = half*8 + (v&3)*2 + ((v>>2)<<4) and +1.
__device__ __forceinline__ void pack_a_store(bf16* __restrict__ base, int m, int k,
                                             int KT, float val) {
  int mt = m >> 4, kt = k >> 5;
  int kk = k & 31;
  int t = kk >> 1, bit = kk & 1;
  int lane = (m & 15) | (((t >> 2) & 1) << 4);
  int v = (t & 3) | ((t >> 3) << 2);
  size_t idx = (((size_t)(mt * KT + kt)) * 256 + lane * 8 + v) * 2 + bit;
  base[idx] = to_bf16(val);
}

// ---------------- weight conversion: f32 row-major (K x N) -> packed-B bf16 --
// Packed-B layout: u32[(kt*NTp + nt)*256 + lane*8 + v] where col n = nt*16+(lane&15),
// dword v holds rows k = kt*32 + (lane>>4)*8 + 2*(v&3) + ((v>>2)<<4) and k+1. Pads n >= Nn with 0.
__global__ __launch_bounds__(256) void pack_b_kernel(const float* __restrict__ in,
                                                     int Kk, int Nn, int NTp,
                                                     bf16* __restrict__ out) {
  size_t idx = (size_t)blockIdx.x * 256 + threadIdx.x;   // one u32 pair per thread
  size_t total = (size_t)(Kk >> 5) * NTp * 256;
  if (idx >= total) return;
  int v = (int)(idx & 7);
  int lane = (int)((idx >> 3) & 31);
  size_t tile = idx >> 8;
  int nt = (int)(tile % NTp), kt = (int)(tile / NTp);
  int n = nt * 16 + (lane & 15);
  int k = kt * 32 + (lane >> 4) * 8 + 2 * (v & 3) + ((v >> 2) << 4);
  float lo = (n < Nn) ? in[(size_t)k * Nn + n] : 0.f;
  float hi = (n < Nn) ? in[(size_t)(k + 1) * Nn + n] : 0.f;
  bf16* o = out + idx * 2;
  o[0] = to_bf16(lo);
  o[1] = to_bf16(hi);
}

// ---------------- WMMA bf16 GEMM (packed operands) with fused epilogues ------
// Per-wave macro tile: 32 (M) x 64 (N) = 2 m-tiles x 4 n-tiles, so each B
// fragment is reused by two WMMAs: 12 b128 loads per 8 v_wmma.
#define EPI_BIAS 1
#define EPI_GELU 2
#define EPI_BN   4
#define EPI_RES  8

union FragU { v16bf v; uint32_t u[8]; uint4 q[2]; };

__global__ __launch_bounds__(256) void gemm_bf16_kernel(
    const uint32_t* __restrict__ Ap,   // packed A: [MT][KT][256]
    const uint32_t* __restrict__ Bp,   // packed B: [KT][NTp][256]
    int M, int Nn, int Kk, int NTp, int epi,
    const float* __restrict__ bias,
    const float* __restrict__ bng, const float* __restrict__ bnb,
    const float* __restrict__ bnm, const float* __restrict__ bnv,
    const float* __restrict__ res, int ldres,
    float* __restrict__ Cf, int ldc,
    bf16*  __restrict__ Cb, int cb_kt) {
  const int lane = threadIdx.x & 31;
  const int wave = threadIdx.x >> 5;
  const int KT = Kk >> 5;
  const int MT2 = M >> 5;               // 32-row macro tiles
  const int NG = NTp >> 2;              // groups of 4 n-tiles
  int wg = blockIdx.x * 8 + wave;
  if (wg >= MT2 * NG) return;
  const int mt2 = wg / NG, ng = wg % NG;
  const int nt0 = ng << 2;
  const int mrow = lane & 15, half = lane >> 4;

  v8f acc[2][4];
#pragma unroll
  for (int im = 0; im < 2; ++im)
#pragma unroll
    for (int j = 0; j < 4; ++j)
      acc[im][j] = (v8f){0.f, 0.f, 0.f, 0.f, 0.f, 0.f, 0.f, 0.f};

  const uint4* aq0 = (const uint4*)Ap + ((size_t)(mt2 * 2) * KT) * 64 + lane * 2;
  const uint4* aq1 = aq0 + (size_t)KT * 64;
  const uint4* bbase = (const uint4*)Bp + lane * 2;

  for (int kt = 0; kt < KT; ++kt) {
    FragU a0, a1, b[4];
    a0.q[0] = aq0[0];
    a0.q[1] = aq0[1];
    a1.q[0] = aq1[0];
    a1.q[1] = aq1[1];
    __builtin_prefetch((const void*)(aq0 + 64), 0, 3);  // next A k-tiles
    __builtin_prefetch((const void*)(aq1 + 64), 0, 3);
    aq0 += 64;
    aq1 += 64;
    const uint4* bq = bbase + ((size_t)kt * NTp + nt0) * 64;
#pragma unroll
    for (int j = 0; j < 4; ++j) {
      b[j].q[0] = bq[j * 64];
      b[j].q[1] = bq[j * 64 + 1];
    }
#pragma unroll
    for (int j = 0; j < 4; ++j) {
      acc[0][j] = __builtin_amdgcn_wmma_f32_16x16x32_bf16(false, a0.v, false, b[j].v,
                                                          (short)0, acc[0][j], false, false);
      acc[1][j] = __builtin_amdgcn_wmma_f32_16x16x32_bf16(false, a1.v, false, b[j].v,
                                                          (short)0, acc[1][j], false, false);
    }
  }

#pragma unroll
  for (int j = 0; j < 4; ++j) {
    int ncol = ((nt0 + j) << 4) + mrow;
    if (ncol >= Nn) continue;
    float bnsc = 0.f, bnof = 0.f, bsv = 0.f;
    if (epi & EPI_BN) {
      bnsc = rsqrtf(bnv[ncol] + 1e-5f) * bng[ncol];
      bnof = bnb[ncol] - bnm[ncol] * bnsc;
    }
    if (epi & EPI_BIAS) bsv = bias[ncol];
#pragma unroll
    for (int im = 0; im < 2; ++im) {
      const int m0 = (mt2 * 2 + im) << 4;   // M is a multiple of 32 for every call
#pragma unroll
      for (int r = 0; r < 8; ++r) {
        int m = m0 + (half << 3) + r;
        float vv = acc[im][j][r];
        if (epi & EPI_BIAS) vv += bsv;
        if (epi & EPI_GELU) vv = gelu_f(vv);
        if (epi & EPI_BN)   vv = vv * bnsc + bnof;
        if (epi & EPI_RES)  vv += res[(size_t)m * ldres + ncol];
        if (Cf) Cf[(size_t)m * ldc + ncol] = vv;
        if (Cb) pack_a_store(Cb, m, ncol, cb_kt, vv);
      }
    }
  }
}

// ---------------- stage 1: neighbor embedding -------------------------------
__global__ __launch_bounds__(256) void nbr_embed_kernel(
    const float* __restrict__ x4, const float* __restrict__ xyz,
    const int* __restrict__ knn,
    const float* __restrict__ W1,
    const float* __restrict__ g1, const float* __restrict__ b1,
    const float* __restrict__ m1, const float* __restrict__ v1,
    const float* __restrict__ W2,
    const float* __restrict__ g2, const float* __restrict__ b2,
    const float* __restrict__ m2, const float* __restrict__ v2,
    const float* __restrict__ W3,
    const float* __restrict__ go, const float* __restrict__ bo,
    const float* __restrict__ mo, const float* __restrict__ vo,
    float* __restrict__ fout, bf16* __restrict__ fbf) {
  int i = blockIdx.x;
  int t = threadIdx.x;
  __shared__ float nbr[KNN_K][8];
  __shared__ float h1[KNN_K][16];
  __shared__ float h2[KNN_K][32];
  if (t < KNN_K) {
    int j = knn[i * KNN_K + t];
    for (int c = 0; c < 3; ++c) nbr[t][c] = xyz[j * 3 + c] - xyz[i * 3 + c];
    for (int c = 0; c < 4; ++c) nbr[t][3 + c] = x4[j * 4 + c];
  }
  __syncthreads();
  {  // h1: 16 neighbors x 16 ch = 256 threads
    int k = t >> 4, c = t & 15;
    float s = 0.f;
    for (int j = 0; j < 7; ++j) s += nbr[k][j] * W1[j * 16 + c];
    s = (s - m1[c]) * rsqrtf(v1[c] + 1e-5f) * g1[c] + b1[c];
    h1[k][c] = gelu_f(s);
  }
  __syncthreads();
  for (int idx = t; idx < KNN_K * 32; idx += 256) {  // h2
    int k = idx >> 5, c = idx & 31;
    float s = 0.f;
    for (int j = 0; j < 16; ++j) s += h1[k][j] * W2[j * 32 + c];
    s = (s - m2[c]) * rsqrtf(v2[c] + 1e-5f) * g2[c] + b2[c];
    h2[k][c] = gelu_f(s);
  }
  __syncthreads();
  {  // e[d] = max_k dot(h2[k], W3[:,d]) then bn_out
    int d = t;
    float mx = -3.0e38f;
    for (int k = 0; k < KNN_K; ++k) {
      float s = 0.f;
      for (int j = 0; j < 32; ++j) s += h2[k][j] * W3[j * DIM + d];
      mx = fmaxf(mx, s);
    }
    float vvv = (mx - mo[d]) * rsqrtf(vo[d] + 1e-5f) * go[d] + bo[d];
    fout[(size_t)i * DIM + d] = vvv;
    pack_a_store(fbf, i, d, DIM / 32, vvv);
  }
}

// ---------------- LFP gather-max + BN + residual -----------------------------
__global__ __launch_bounds__(256) void lfp_gather_kernel(
    const float* __restrict__ fproj, const int* __restrict__ knn,
    const float* __restrict__ g, const float* __restrict__ b,
    const float* __restrict__ m, const float* __restrict__ v,
    float* __restrict__ x, bf16* __restrict__ xbf) {
  int i = blockIdx.x, d = threadIdx.x;
  float mx = -3.0e38f;
  for (int k = 0; k < KNN_K; ++k) {
    int j = knn[i * KNN_K + k];
    mx = fmaxf(mx, fproj[(size_t)j * DIM + d]);
  }
  float gg = mx - fproj[(size_t)i * DIM + d];
  gg = (gg - m[d]) * rsqrtf(v[d] + 1e-5f) * g[d] + b[d];
  float o = x[(size_t)i * DIM + d] + gg;
  x[(size_t)i * DIM + d] = o;
  pack_a_store(xbf, i, d, DIM / 32, o);
}

// ---------------- RMS norm (D=256) -> packed bf16 ----------------------------
__global__ __launch_bounds__(256) void rms256_kernel(const float* __restrict__ x,
                                                     const float* __restrict__ w,
                                                     bf16* __restrict__ out) {
  int i = blockIdx.x, t = threadIdx.x;
  float vv = x[(size_t)i * DIM + t];
  __shared__ float red[256];
  red[t] = vv * vv;
  __syncthreads();
  for (int s = 128; s > 0; s >>= 1) {
    if (t < s) red[t] += red[t + s];
    __syncthreads();
  }
  float sc = rsqrtf(red[0] / (float)DIM + 1e-5f);
  pack_a_store(out, i, t, DIM / 32, vv * sc * w[t]);
}

// ---------------- conv1d(K=4 causal) + silu + dt/dA --------------------------
__global__ __launch_bounds__(256) void conv_dt_kernel(
    const float* __restrict__ zx, const float* __restrict__ cw,
    const float* __restrict__ cb, const float* __restrict__ dtb,
    const float* __restrict__ alog,
    float* __restrict__ xbc, float* __restrict__ dto, float* __restrict__ dao) {
  int i = blockIdx.x, t = threadIdx.x;
  for (int c = t; c < XBCC; c += 256) {
    float s = cb[c];
    for (int j = 0; j < 4; ++j) {
      int r = i - 3 + j;
      if (r >= 0) s += zx[(size_t)r * ZLD + DIN + c] * cw[c * 4 + j];
    }
    xbc[(size_t)i * XBCC + c] = silu_f(s);
  }
  if (t < NHEADS) {
    float d = zx[(size_t)i * ZLD + (ZCOLS - NHEADS) + t] + dtb[t];
    d = (d > 20.f) ? d : log1pf(__expf(d));  // softplus
    dto[i * NHEADS + t] = d;
    dao[i * NHEADS + t] = __expf(-__expf(alog[t]) * d);
  }
}

// ---------------- chunked parallel scan --------------------------------------
// pass 1: per-chunk local end state + cumulative decay
__global__ __launch_bounds__(256) void scan_pass1(
    const float* __restrict__ xbc, const float* __restrict__ dtb,
    const float* __restrict__ dab, float* __restrict__ Sc,
    float* __restrict__ Aprod) {
  int c = blockIdx.x >> 3, h = blockIdx.x & 7;
  int t = threadIdx.x;
  int base = t * 16, p0 = base >> 6, s0 = base & 63;
  float S[16];
#pragma unroll
  for (int e = 0; e < 16; ++e) S[e] = 0.f;
  __shared__ float shx[64], shB[64];
  __shared__ float sha, shdt;
  float aprod = 1.f;
  for (int ii = 0; ii < LCHUNK; ++ii) {
    int i = c * LCHUNK + ii;
    if (t < 64) shx[t] = xbc[(size_t)i * XBCC + h * HEADD + t];
    else if (t < 128) shB[t - 64] = xbc[(size_t)i * XBCC + DIN + (t - 64)];
    else if (t == 128) sha = dab[i * NHEADS + h];
    else if (t == 129) shdt = dtb[i * NHEADS + h];
    __syncthreads();
    float a = sha;
    float xdt = shx[p0] * shdt;
#pragma unroll
    for (int e = 0; e < 16; ++e) S[e] = S[e] * a + xdt * shB[s0 + e];
    aprod *= a;
    __syncthreads();
  }
  float* out = Sc + (size_t)blockIdx.x * 4096 + base;
#pragma unroll
  for (int e = 0; e < 16; ++e) out[e] = S[e];
  if (t == 0) Aprod[blockIdx.x] = aprod;
}

// pass 2: sequential inter-chunk combine (only 128 steps)
__global__ __launch_bounds__(256) void scan_pass2(const float* __restrict__ Sc,
                                                  const float* __restrict__ Aprod,
                                                  float* __restrict__ Hinit) {
  int h = blockIdx.x >> 4;
  int elem = (blockIdx.x & 15) * 256 + threadIdx.x;
  float H = 0.f;
  for (int c = 0; c < NCHUNK; ++c) {
    size_t idx = ((size_t)(c * NHEADS + h)) * 4096 + elem;
    Hinit[idx] = H;
    H = H * Aprod[c * NHEADS + h] + Sc[idx];
  }
}

// pass 3: replay chunk with true initial state, emit y (+ D*xs)
__global__ __launch_bounds__(64) void scan_pass3(
    const float* __restrict__ xbc, const float* __restrict__ dtb,
    const float* __restrict__ dab, const float* __restrict__ Hinit,
    const float* __restrict__ Dp, float* __restrict__ y) {
  int c = blockIdx.x >> 3, h = blockIdx.x & 7;
  int p = threadIdx.x;  // headdim row
  float St[64];
  const float* Hi = Hinit + ((size_t)(c * NHEADS + h)) * 4096 + p * 64;
#pragma unroll
  for (int s = 0; s < 64; ++s) St[s] = Hi[s];
  __shared__ float shB[64], shC[64], shx[64];
  __shared__ float sha, shdt;
  float Dh = Dp[h];
  for (int ii = 0; ii < LCHUNK; ++ii) {
    int i = c * LCHUNK + ii;
    shB[p] = xbc[(size_t)i * XBCC + DIN + p];
    shC[p] = xbc[(size_t)i * XBCC + DIN + DSTATE + p];
    shx[p] = xbc[(size_t)i * XBCC + h * HEADD + p];
    if (p == 0) sha = dab[i * NHEADS + h];
    if (p == 1) shdt = dtb[i * NHEADS + h];
    __syncthreads();
    float a = sha, xdt = shx[p] * shdt;
    float accv = 0.f;
#pragma unroll
    for (int s = 0; s < 64; ++s) {
      St[s] = St[s] * a + xdt * shB[s];
      accv += St[s] * shC[s];
    }
    y[(size_t)i * DIN + h * HEADD + p] = accv + Dh * shx[p];
    __syncthreads();
  }
}

// ---------------- gate (y*silu(z)) + RMS(512) -> packed bf16 -----------------
__global__ __launch_bounds__(256) void gate_rms_kernel(
    const float* __restrict__ y, const float* __restrict__ zx,
    const float* __restrict__ w, bf16* __restrict__ out) {
  int i = blockIdx.x, t = threadIdx.x;
  float a0 = y[(size_t)i * DIN + t] * silu_f(zx[(size_t)i * ZLD + t]);
  float a1 = y[(size_t)i * DIN + 256 + t] * silu_f(zx[(size_t)i * ZLD + 256 + t]);
  __shared__ float red[256];
  red[t] = a0 * a0 + a1 * a1;
  __syncthreads();
  for (int s = 128; s > 0; s >>= 1) {
    if (t < s) red[t] += red[t + s];
    __syncthreads();
  }
  float sc = rsqrtf(red[0] / (float)DIN + 1e-5f);
  pack_a_store(out, i, t, DIN / 32, a0 * sc * w[t]);
  pack_a_store(out, i, t + 256, DIN / 32, a1 * sc * w[t + 256]);
}

// ---------------- BN -> packed bf16 (post head input) ------------------------
__global__ __launch_bounds__(256) void bn_bf16_kernel(
    const float* __restrict__ x, const float* __restrict__ g,
    const float* __restrict__ b, const float* __restrict__ m,
    const float* __restrict__ v, bf16* __restrict__ out) {
  int i = blockIdx.x, d = threadIdx.x;
  float vv = x[(size_t)i * DIM + d];
  vv = (vv - m[d]) * rsqrtf(v[d] + 1e-5f) * g[d] + b[d];
  pack_a_store(out, i, d, DIM / 32, vv);
}

// ============================ host orchestration =============================
enum {
  IN_X = 0, IN_XYZ, IN_KNN,
  NBR_W1, NBR_BN1_G, NBR_BN1_B, NBR_BN1_M, NBR_BN1_V,
  NBR_W2, NBR_BN2_G, NBR_BN2_B, NBR_BN2_M, NBR_BN2_V,
  NBR_W3, NBR_BNO_G, NBR_BNO_B, NBR_BNO_M, NBR_BNO_V,
  MLP0_W1, MLP0_B1, MLP0_W2, MLP0_BN_G, MLP0_BN_B, MLP0_BN_M, MLP0_BN_V,
  LFP_BASE,                               // 25: 4 x {proj, g, b, m, v}
  MLPS0_W1 = LFP_BASE + 20, MLPS0_B1, MLPS0_W2, MLPS0_BN_G, MLPS0_BN_B, MLPS0_BN_M, MLPS0_BN_V,
  MLPS1_W1, MLPS1_B1, MLPS1_W2, MLPS1_BN_G, MLPS1_BN_B, MLPS1_BN_M, MLPS1_BN_V,
  MB_INPROJ, MB_CONVW, MB_CONVB, MB_DTBIAS, MB_ALOG, MB_D, MB_NORMW, MB_PREW, MB_OUTPROJ,
  POST_BN_G, POST_BN_B, POST_BN_M, POST_BN_V, POST_W
};

static void launch_gemm(hipStream_t s, const bf16* A, const bf16* B,
                        int M, int Nn, int Kk, int NTp, int epi, const float* bias,
                        const float* g, const float* b, const float* m, const float* v,
                        const float* res, int ldres, float* Cf, int ldc,
                        bf16* Cb, int cb_kt) {
  int wgs = (M >> 5) * (NTp >> 2);
  int blocks = (wgs + 7) / 8;
  gemm_bf16_kernel<<<blocks, 256, 0, s>>>((const uint32_t*)A, (const uint32_t*)B,
                                          M, Nn, Kk, NTp, epi, bias, g, b, m, v,
                                          res, ldres, Cf, ldc, Cb, cb_kt);
}

extern "C" void kernel_launch(void* const* d_in, const int* in_sizes, int n_in,
                              void* d_out, int out_size, void* d_ws, size_t ws_size,
                              hipStream_t stream) {
  (void)in_sizes; (void)n_in; (void)out_size; (void)ws_size;
  auto F = [&](int i) { return (const float*)d_in[i]; };
  const int* knn = (const int*)d_in[IN_KNN];

  // -------- workspace carve-out --------
  size_t off = 0;
  auto alloc = [&](size_t bytes) {
    off = (off + 255) & ~(size_t)255;
    void* p = (char*)d_ws + off;
    off += bytes;
    return p;
  };
  float* xf    = (float*)alloc((size_t)N_PTS * DIM * 4);
  float* fproj = (float*)alloc((size_t)N_PTS * DIM * 4);
  float* ybuf  = (float*)alloc((size_t)N_PTS * DIN * 4);
  float* zx    = (float*)alloc((size_t)N_PTS * ZLD * 4);
  float* xbc   = (float*)alloc((size_t)N_PTS * XBCC * 4);
  float* dtb   = (float*)alloc((size_t)N_PTS * NHEADS * 4);
  float* dab   = (float*)alloc((size_t)N_PTS * NHEADS * 4);
  float* Sc    = (float*)alloc((size_t)NCHUNK * NHEADS * 4096 * 4);
  float* Hinit = (float*)alloc((size_t)NCHUNK * NHEADS * 4096 * 4);
  float* Aprod = (float*)alloc((size_t)NCHUNK * NHEADS * 4);
  bf16* xbf    = (bf16*)alloc((size_t)N_PTS * DIM * 2);   // packed-A, KT=8
  bf16* xrbf   = (bf16*)alloc((size_t)N_PTS * DIM * 2);   // packed-A, KT=8
  bf16* hbf    = (bf16*)alloc((size_t)N_PTS * DIN * 2);   // packed-A, KT=16

  // packed-B weights: bytes = (K/32) * NTp * 256 * 4
  auto allocB = [&](int Kk, int NTp) {
    return (bf16*)alloc((size_t)(Kk >> 5) * NTp * 1024);
  };
  bf16 *w_mlp1[3], *w_mlp2[3];
  for (int i = 0; i < 3; ++i) {
    w_mlp1[i] = allocB(DIM, 32);    // 256 -> 512
    w_mlp2[i] = allocB(HIDD, 16);   // 512 -> 256
  }
  bf16* w_lfp[4];
  for (int i = 0; i < 4; ++i) w_lfp[i] = allocB(DIM, 16);
  bf16* w_inp  = allocB(DIM, 76);   // 256 -> 1160 (padded to 76 n-tiles)
  bf16* w_outp = allocB(DIN, 16);   // 512 -> 256
  bf16* w_post = allocB(DIM, 16);   // 256 -> 256

  // -------- weight packing --------
  auto packB = [&](const float* in, int Kk, int Nn, int NTp, bf16* out) {
    int blocks = (Kk >> 5) * NTp;   // one block per 256-u32 tile
    pack_b_kernel<<<blocks, 256, 0, stream>>>(in, Kk, Nn, NTp, out);
  };
  const int mlpw1[3] = {MLP0_W1, MLPS0_W1, MLPS1_W1};
  const int mlpw2[3] = {MLP0_W2, MLPS0_W2, MLPS1_W2};
  for (int i = 0; i < 3; ++i) {
    packB(F(mlpw1[i]), DIM, HIDD, 32, w_mlp1[i]);
    packB(F(mlpw2[i]), HIDD, DIM, 16, w_mlp2[i]);
  }
  for (int i = 0; i < 4; ++i) packB(F(LFP_BASE + 5 * i), DIM, DIM, 16, w_lfp[i]);
  packB(F(MB_INPROJ), DIM, ZCOLS, 76, w_inp);
  packB(F(MB_OUTPROJ), DIN, DIM, 16, w_outp);
  packB(F(POST_W), DIM, DIM, 16, w_post);

  // -------- stage 1: neighbor embedding -> xf, xbf --------
  nbr_embed_kernel<<<N_PTS, 256, 0, stream>>>(
      F(IN_X), F(IN_XYZ), knn, F(NBR_W1),
      F(NBR_BN1_G), F(NBR_BN1_B), F(NBR_BN1_M), F(NBR_BN1_V), F(NBR_W2),
      F(NBR_BN2_G), F(NBR_BN2_B), F(NBR_BN2_M), F(NBR_BN2_V), F(NBR_W3),
      F(NBR_BNO_G), F(NBR_BNO_B), F(NBR_BNO_M), F(NBR_BNO_V), xf, xbf);

  // -------- MLP helper --------
  auto run_mlp = [&](int wi, const float* b1, const float* g, const float* b,
                     const float* m, const float* v) {
    launch_gemm(stream, xbf, w_mlp1[wi], N_PTS, HIDD, DIM, 32,
                EPI_BIAS | EPI_GELU, b1, 0, 0, 0, 0, 0, 0, nullptr, 0, hbf, DIN / 32);
    launch_gemm(stream, hbf, w_mlp2[wi], N_PTS, DIM, HIDD, 16,
                EPI_BN | EPI_RES, 0, g, b, m, v, xf, DIM, xf, DIM, xbf, DIM / 32);
  };

  // x = x + mlp0(x)
  run_mlp(0, F(MLP0_B1), F(MLP0_BN_G), F(MLP0_BN_B), F(MLP0_BN_M), F(MLP0_BN_V));

  for (int it = 0; it < 4; ++it) {
    // ---- LFP ----
    int lb = LFP_BASE + 5 * it;
    launch_gemm(stream, xbf, w_lfp[it], N_PTS, DIM, DIM, 16, 0,
                0, 0, 0, 0, 0, 0, 0, fproj, DIM, nullptr, 0);
    lfp_gather_kernel<<<N_PTS, 256, 0, stream>>>(fproj, knn, F(lb + 1), F(lb + 2),
                                                 F(lb + 3), F(lb + 4), xf, xbf);
    // ---- Mamba2 ----
    rms256_kernel<<<N_PTS, 256, 0, stream>>>(xf, F(MB_PREW), xrbf);
    launch_gemm(stream, xrbf, w_inp, N_PTS, ZCOLS, DIM, 76, 0,
                0, 0, 0, 0, 0, 0, 0, zx, ZLD, nullptr, 0);
    conv_dt_kernel<<<N_PTS, 256, 0, stream>>>(zx, F(MB_CONVW), F(MB_CONVB),
                                              F(MB_DTBIAS), F(MB_ALOG), xbc, dtb, dab);
    scan_pass1<<<NCHUNK * NHEADS, 256, 0, stream>>>(xbc, dtb, dab, Sc, Aprod);
    scan_pass2<<<NHEADS * 16, 256, 0, stream>>>(Sc, Aprod, Hinit);
    scan_pass3<<<NCHUNK * NHEADS, 64, 0, stream>>>(xbc, dtb, dab, Hinit, F(MB_D), ybuf);
    gate_rms_kernel<<<N_PTS, 256, 0, stream>>>(ybuf, zx, F(MB_NORMW), hbf);
    launch_gemm(stream, hbf, w_outp, N_PTS, DIM, DIN, 16, EPI_RES,
                0, 0, 0, 0, 0, xf, DIM, xf, DIM, xbf, DIM / 32);
    // ---- interleaved MLPs ----
    if (it == 1)
      run_mlp(1, F(MLPS0_B1), F(MLPS0_BN_G), F(MLPS0_BN_B), F(MLPS0_BN_M), F(MLPS0_BN_V));
    if (it == 3)
      run_mlp(2, F(MLPS1_B1), F(MLPS1_BN_G), F(MLPS1_BN_B), F(MLPS1_BN_M), F(MLPS1_BN_V));
  }

  // -------- post head: bn -> packed bf16 -> GEMM to d_out --------
  bn_bf16_kernel<<<N_PTS, 256, 0, stream>>>(xf, F(POST_BN_G), F(POST_BN_B),
                                            F(POST_BN_M), F(POST_BN_V), xbf);
  launch_gemm(stream, xbf, w_post, N_PTS, DIM, DIM, 16, 0,
              0, 0, 0, 0, 0, 0, 0, (float*)d_out, DIM, nullptr, 0);
}
